// FissaNetwork_39994735460682
// MI455X (gfx1250) — matrix operand order repaired and checked by reading, not compiled, not run on hardware
//
#include <hip/hip_runtime.h>
#include <hip/hip_bf16.h>
#include <stdint.h>

// ---------------------------------------------------------------------------
// FISSA forward for MI455X (gfx1250, wave32).
// GEMMs run on v_wmma_f32_16x16x32_bf16 (f32 accumulate); softmax / layernorm /
// gating stay in f32 VALU. Global branch is algebraically reduced to GEMVs.
// ---------------------------------------------------------------------------

typedef __attribute__((ext_vector_type(16))) __bf16 v16bf;
typedef __attribute__((ext_vector_type(8)))  float  v8f;

namespace {
constexpr int   kB   = 512;
constexpr int   kL   = 200;
constexpr int   kU   = 128;
constexpr int   kH   = 2;
constexpr int   kDH  = 64;
constexpr int   kLP  = 224;                // padded length, multiple of 32 (7x32, 14x16)
constexpr float kNEG = -4294967295.0f;
constexpr float kEPS = 1e-8f;
}

// ---------------- wave32 helpers ----------------
__device__ inline float wave_sum(float v) {
#pragma unroll
  for (int off = 16; off > 0; off >>= 1) v += __shfl_xor(v, off, 32);
  return v;
}

// ---------------- WMMA fragment helpers (CDNA5 wave32 layouts) -------------
// A: 16x32 bf16 tile, row stride lda.  lane = {m = lane&15, k-base by lane half}
// elements 0..7  -> K = 8*(lane>>4) + e
// elements 8..15 -> K = 16 + 8*(lane>>4) + (e-8)
__device__ inline v16bf frag_a(const __bf16* A, int lda) {
  const int lane = threadIdx.x & 31;
  const int m    = lane & 15;
  const int kb   = (lane >> 4) << 3;
  const __bf16* r = A + m * lda;
  v16bf a;
#pragma unroll
  for (int e = 0; e < 8; ++e) a[e] = r[kb + e];
#pragma unroll
  for (int e = 0; e < 8; ++e) a[8 + e] = r[16 + kb + e];
  return a;
}

// B: 32x16 bf16 tile, row stride ldb.  lane = {n = lane&15, k-base by lane half}, K order as frag_a.
__device__ inline v16bf frag_b(const __bf16* Bm, int ldb) {
  const int lane = threadIdx.x & 31;
  const int n    = lane & 15;
  const int kb   = (lane >> 4) << 3;
  const __bf16* r = Bm + n;
  v16bf b;
#pragma unroll
  for (int e = 0; e < 8; ++e) b[e] = r[(kb + e) * ldb];
#pragma unroll
  for (int e = 0; e < 8; ++e) b[8 + e] = r[(16 + kb + e) * ldb];
  return b;
}

__device__ inline v8f wmma_bf16(v16bf a, v16bf b, v8f c) {
  return __builtin_amdgcn_wmma_f32_16x16x32_bf16(
      /*neg_a=*/false, a, /*neg_b=*/false, b,
      /*c_mod=*/(short)0, c, /*reuse_a=*/false, /*reuse_b=*/false);
}

// C/D layout: n = lane&15 ; m = 8*(lane>>4) + j
__device__ inline int c_n() { return threadIdx.x & 15; }
__device__ inline int c_m() { return ((threadIdx.x & 31) >> 4) << 3; }

// ---------------- row layernorm (wave per row, 4 ch / lane) ----------------
__device__ inline void ln_write(float x[4], float* dst, int lane) {
  float s = x[0] + x[1] + x[2] + x[3];
  float mu = wave_sum(s) * (1.0f / kU);
  float v = 0.f;
#pragma unroll
  for (int j = 0; j < 4; ++j) { float d = x[j] - mu; v += d * d; }
  v = wave_sum(v) * (1.0f / kU);
  float inv = rsqrtf(v + kEPS);
#pragma unroll
  for (int j = 0; j < 4; ++j) dst[lane * 4 + j] = (x[j] - mu) * inv;
}

// ---------------- kernels --------------------------------------------------

// fp32 -> bf16 weight conversion
__global__ __launch_bounds__(256) void k_cvt(const float* __restrict__ src,
                                             __bf16* __restrict__ dst, int n) {
  int i = blockIdx.x * blockDim.x + threadIdx.x;
  if (i < n) dst[i] = (__bf16)src[i];
}

// inputs = LN((item_emb[user] + pos_emb) * mask)
__global__ __launch_bounds__(256) void k_embed(const float* __restrict__ item_emb,
                                               const float* __restrict__ pos_emb,
                                               const int* __restrict__ user_items,
                                               const float* __restrict__ mask,
                                               float* __restrict__ out) {
  int row = blockIdx.x * 8 + (threadIdx.x >> 5);
  if (row >= kB * kL) return;
  int lane = threadIdx.x & 31;
  int l = row % kL;
  int idx = user_items[row];
  float mk = mask[row];
  float x[4];
#pragma unroll
  for (int j = 0; j < 4; ++j) {
    int c = lane * 4 + j;
    x[j] = (item_emb[(size_t)idx * kU + c] + pos_emb[l * kU + c]) * mk;
  }
  ln_write(x, out + (size_t)row * kU, lane);
}

// generic rowwise (optional mask) layernorm; safe in-place
__global__ __launch_bounds__(256) void k_ln(const float* __restrict__ in,
                                            float* __restrict__ out,
                                            const float* __restrict__ mask,
                                            int nrows) {
  int row = blockIdx.x * 8 + (threadIdx.x >> 5);
  if (row >= nrows) return;
  int lane = threadIdx.x & 31;
  float mk = mask ? mask[row] : 1.0f;
  float x[4];
#pragma unroll
  for (int j = 0; j < 4; ++j) x[j] = in[(size_t)row * kU + lane * 4 + j] * mk;
  ln_write(x, out + (size_t)row * kU, lane);
}

// Q/K/V projection: X[b, m0..m0+15, :] @ {Wq,Wk,Wv}.  K stored transposed.
// grid = B * (kLP/16), block = 256 (8 waves; wave w owns N-tile w)
__global__ __launch_bounds__(256) void k_qkv(const float* __restrict__ X,
                                             const __bf16* __restrict__ Wq,
                                             const __bf16* __restrict__ Wk,
                                             const __bf16* __restrict__ Wv,
                                             __bf16* __restrict__ Q,
                                             __bf16* __restrict__ Kt,
                                             __bf16* __restrict__ V) {
  __shared__ __align__(16) __bf16 Xs[16 * kU];
  int b  = blockIdx.x / (kLP / 16);
  int mt = blockIdx.x % (kLP / 16);
  int m0 = mt * 16;
  for (int i = threadIdx.x; i < 16 * kU; i += blockDim.x) {
    int r = i >> 7, u = i & 127;
    int l = m0 + r;
    Xs[i] = (__bf16)((l < kL) ? X[((size_t)b * kL + l) * kU + u] : 0.0f);
  }
  __syncthreads();
  int n0 = (threadIdx.x >> 5) * 16;
  int cn = c_n(), cm = c_m();
  const __bf16* Ws[3] = {Wq, Wk, Wv};
#pragma unroll
  for (int t = 0; t < 3; ++t) {
    v8f c = {};
#pragma unroll
    for (int k0 = 0; k0 < kU; k0 += 32)
      c = wmma_bf16(frag_a(Xs + k0, kU), frag_b(Ws[t] + k0 * kU + n0, kU), c);
#pragma unroll
    for (int j = 0; j < 8; ++j) {
      int m = m0 + cm + j;
      int n = n0 + cn;
      __bf16 hv = (__bf16)c[j];
      if (t == 0)      Q[((size_t)b * kLP + m) * kU + n]  = hv;
      else if (t == 1) Kt[((size_t)b * kU + n) * kLP + m] = hv;
      else             V[((size_t)b * kLP + m) * kU + n]  = hv;
    }
  }
}

// fused attention for one (b, head, 16-row query tile):
// S = Qh Kh^T / 8, causal+key mask, softmax (*query mask), O = P Vh, +residual.
// grid = B * H * 13, block = 32 (one wave)
__global__ __launch_bounds__(32) void k_attn(const __bf16* __restrict__ Q,
                                             const __bf16* __restrict__ Kt,
                                             const __bf16* __restrict__ V,
                                             const float* __restrict__ X,
                                             const float* __restrict__ mask,
                                             float* __restrict__ out) {
  __shared__ __align__(16) float  Sf[16 * kLP];
  __shared__ __align__(16) __bf16 Pb[16 * kLP];
  int id = blockIdx.x;
  int mt = id % 13; id /= 13;
  int h  = id % kH;
  int b  = id / kH;
  int q0 = mt * 16;
  int lane = threadIdx.x;
  const __bf16* Qb  = Q  + (size_t)b * kLP * kU;
  const __bf16* Ktb = Kt + (size_t)b * kU * kLP;
  const __bf16* Vb  = V  + (size_t)b * kLP * kU;
  int cn = c_n(), cm = c_m();

  // -------- scores --------
  for (int nt = 0; nt < 13; ++nt) {
    v8f c = {};
#pragma unroll
    for (int k0 = 0; k0 < kDH; k0 += 32) {
      v16bf a  = frag_a(Qb + q0 * kU + h * kDH + k0, kU);
      v16bf bf = frag_b(Ktb + (h * kDH + k0) * kLP + nt * 16, kLP);
      c = wmma_bf16(a, bf, c);
    }
    int col = nt * 16 + cn;
    float km = (col < kL) ? mask[b * kL + col] : 0.0f;
#pragma unroll
    for (int j = 0; j < 8; ++j) {
      int row = q0 + cm + j;
      float v = c[j] * 0.125f;                 // 1/sqrt(64)
      if (col > row || km == 0.0f) v = kNEG;   // causal + key padding mask
      Sf[(cm + j) * kLP + col] = v;
    }
  }
  __syncthreads();

  // -------- softmax (lanes 0..15, one row each) --------
  if (lane < 16) {
    int row = q0 + lane;
    float* sr = Sf + lane * kLP;
    float mx = -3.4e38f;
    for (int cpos = 0; cpos < 208; ++cpos) mx = fmaxf(mx, sr[cpos]);
    float sum = 0.f;
    for (int cpos = 0; cpos < 208; ++cpos) {
      float e = __expf(sr[cpos] - mx);
      sr[cpos] = e;
      sum += e;
    }
    float qm  = (row < kL) ? mask[b * kL + row] : 0.0f;
    float inv = qm / sum;
    __bf16* pr = Pb + lane * kLP;
    for (int cpos = 0; cpos < 208; ++cpos) pr[cpos] = (__bf16)(sr[cpos] * inv);
    for (int cpos = 208; cpos < kLP; ++cpos) pr[cpos] = (__bf16)0.0f;
  }
  __syncthreads();

  // -------- O = P @ Vh (+ residual) --------
  for (int nt = 0; nt < kDH / 16; ++nt) {
    v8f c = {};
#pragma unroll
    for (int k0 = 0; k0 < kLP; k0 += 32) {
      v16bf a  = frag_a(Pb + k0, kLP);
      v16bf bf = frag_b(Vb + k0 * kU + h * kDH + nt * 16, kU);
      c = wmma_bf16(a, bf, c);
    }
#pragma unroll
    for (int j = 0; j < 8; ++j) {
      int row = q0 + cm + j;
      if (row < kL) {
        size_t ix = ((size_t)b * kL + row) * kU + h * kDH + nt * 16 + cn;
        out[ix] = c[j] + X[ix];
      }
    }
  }
}

// FFN: out = Y + relu(Y@W1 + b1)@W2 + b2, per 16-row tile
// grid = B * 13, block = 256 (8 waves, wave w owns N-tile w)
__global__ __launch_bounds__(256) void k_ffn(const float* __restrict__ Y,
                                             const __bf16* __restrict__ W1,
                                             const __bf16* __restrict__ W2,
                                             const float* __restrict__ b1,
                                             const float* __restrict__ b2,
                                             float* __restrict__ out) {
  __shared__ __align__(16) __bf16 Ys[16 * kU];
  __shared__ __align__(16) __bf16 H1s[16 * kU];
  int b  = blockIdx.x / 13;
  int mt = blockIdx.x % 13;
  int m0 = mt * 16;
  for (int i = threadIdx.x; i < 16 * kU; i += blockDim.x) {
    int r = i >> 7, u = i & 127;
    int l = m0 + r;
    Ys[i] = (__bf16)((l < kL) ? Y[((size_t)b * kL + l) * kU + u] : 0.0f);
  }
  __syncthreads();
  int n0 = (threadIdx.x >> 5) * 16;
  int cn = c_n(), cm = c_m();

  v8f c = {};
#pragma unroll
  for (int k0 = 0; k0 < kU; k0 += 32)
    c = wmma_bf16(frag_a(Ys + k0, kU), frag_b(W1 + k0 * kU + n0, kU), c);
  float bias1 = b1[n0 + cn];
#pragma unroll
  for (int j = 0; j < 8; ++j)
    H1s[(cm + j) * kU + n0 + cn] = (__bf16)fmaxf(c[j] + bias1, 0.0f);
  __syncthreads();

  v8f c2 = {};
#pragma unroll
  for (int k0 = 0; k0 < kU; k0 += 32)
    c2 = wmma_bf16(frag_a(H1s + k0, kU), frag_b(W2 + k0 * kU + n0, kU), c2);
  float bias2 = b2[n0 + cn];
#pragma unroll
  for (int j = 0; j < 8; ++j) {
    int l = m0 + cm + j;
    if (l < kL) {
      size_t ix = ((size_t)b * kL + l) * kU + n0 + cn;
      out[ix] = Y[ix] + c2[j] + bias2;
    }
  }
}

// r_h = Wk[:, h*64:(h+1)*64] @ q_h  (folds loc-attn score GEMM into 2 vectors)
__global__ __launch_bounds__(128) void k_locq(const float* __restrict__ Wk,
                                              const float* __restrict__ Qv,
                                              float* __restrict__ rq) {
  int u = threadIdx.x;
  float r0 = 0.f, r1 = 0.f;
  for (int d = 0; d < kDH; ++d) {
    r0 += Wk[u * kU + d] * Qv[d];
    r1 += Wk[u * kU + kDH + d] * Qv[kDH + d];
  }
  rq[u] = r0;
  rq[kU + u] = r1;
}

// location attention per user: s = E r_h, softmax, ebar = p^T E, o = ebar Wv
// grid = B, block = 256
__global__ __launch_bounds__(256) void k_glo_attn(const float* __restrict__ item_emb,
                                                  const int* __restrict__ user_items,
                                                  const float* __restrict__ mask,
                                                  const float* __restrict__ rq,
                                                  const float* __restrict__ Wv,
                                                  float* __restrict__ glo_raw) {
  __shared__ float sc[2][208];
  __shared__ float ebar[2][kU];
  int b = blockIdx.x, tid = threadIdx.x;
  for (int m = tid; m < kL; m += 256) {
    int idx = user_items[b * kL + m];
    const float* e = item_emb + (size_t)idx * kU;
    float s0 = 0.f, s1 = 0.f;
    for (int u = 0; u < kU; ++u) {
      float ev = e[u];
      s0 += ev * rq[u];
      s1 += ev * rq[kU + u];
    }
    float mk = mask[b * kL + m];
    sc[0][m] = (mk == 0.f) ? kNEG : s0;
    sc[1][m] = (mk == 0.f) ? kNEG : s1;
  }
  __syncthreads();
  if (tid < 2) {
    float mx = -3.4e38f;
    for (int m = 0; m < kL; ++m) mx = fmaxf(mx, sc[tid][m]);
    float sum = 0.f;
    for (int m = 0; m < kL; ++m) { float e = __expf(sc[tid][m] - mx); sc[tid][m] = e; sum += e; }
    float inv = 1.0f / sum;
    for (int m = 0; m < kL; ++m) sc[tid][m] *= inv;
  }
  __syncthreads();
  {
    int u = tid & 127, h = tid >> 7;
    float acc = 0.f;
    for (int m = 0; m < kL; ++m) {
      int idx = user_items[b * kL + m];
      acc += sc[h][m] * item_emb[(size_t)idx * kU + u];
    }
    ebar[h][u] = acc;
  }
  __syncthreads();
  if (tid < kU) {
    int h = tid / kDH;   // output column tid = h*64 + d -> uses head-h probs
    float acc = 0.f;
    for (int u = 0; u < kU; ++u) acc += ebar[h][u] * Wv[u * kU + tid];
    glo_raw[b * kU + tid] = acc;
  }
}

__device__ inline float block_sum128(float v, float* red) {
  v = wave_sum(v);
  if ((threadIdx.x & 31) == 0) red[threadIdx.x >> 5] = v;
  __syncthreads();
  float t = red[0] + red[1] + red[2] + red[3];
  __syncthreads();
  return t;
}

// global branch post: glo = LN( LN(glo_raw) + FFN(LN(glo_raw)) )
// grid = B, block = 128
__global__ __launch_bounds__(128) void k_glo_post(const float* __restrict__ glo_raw,
                                                  const float* __restrict__ W1,
                                                  const float* __restrict__ bb1,
                                                  const float* __restrict__ W2,
                                                  const float* __restrict__ bb2,
                                                  float* __restrict__ glo) {
  __shared__ float y[kU];
  __shared__ float h1[kU];
  __shared__ float red[4];
  int b = blockIdx.x, u = threadIdx.x;
  float x = glo_raw[b * kU + u];
  float mu = block_sum128(x, red) * (1.0f / kU);
  float d = x - mu;
  float var = block_sum128(d * d, red) * (1.0f / kU);
  float yl = d * rsqrtf(var + kEPS);
  y[u] = yl;
  __syncthreads();
  float a1 = bb1[u];
  for (int u2 = 0; u2 < kU; ++u2) a1 += y[u2] * W1[u2 * kU + u];
  h1[u] = fmaxf(a1, 0.0f);
  __syncthreads();
  float a2 = bb2[u];
  for (int u2 = 0; u2 < kU; ++u2) a2 += h1[u2] * W2[u2 * kU + u];
  float z = yl + a2;
  mu = block_sum128(z, red) * (1.0f / kU);
  d = z - mu;
  var = block_sum128(d * d, red) * (1.0f / kU);
  glo[b * kU + u] = d * rsqrtf(var + kEPS);
}

// gating + LN + logits for 2B*L candidate rows (wave per row)
__global__ __launch_bounds__(256) void k_gate(const float* __restrict__ loc,
                                              const float* __restrict__ glo,
                                              const float* __restrict__ item_emb,
                                              const int* __restrict__ user_items,
                                              const int* __restrict__ pos_items,
                                              const int* __restrict__ neg_items,
                                              const float* __restrict__ gw,
                                              const float* __restrict__ gb,
                                              const float* __restrict__ mask,
                                              float* __restrict__ out) {
  int row = blockIdx.x * 8 + (threadIdx.x >> 5);
  if (row >= 2 * kB * kL) return;
  int lane = threadIdx.x & 31;
  int s = row / (kB * kL);
  int r = row % (kB * kL);          // = b*L + l
  int b = r / kL;
  const int* items = s ? neg_items : pos_items;
  int ci = items[r];
  int ui = user_items[r];
  float lv[4], gv[4], iv[4], cv[4];
  float dot = 0.f;
#pragma unroll
  for (int j = 0; j < 4; ++j) {
    int c = lane * 4 + j;
    gv[j] = glo[b * kU + c];
    iv[j] = item_emb[(size_t)ui * kU + c];
    cv[j] = item_emb[(size_t)ci * kU + c];
    lv[j] = loc[(size_t)r * kU + c];
    dot += gv[j] * gw[c] + iv[j] * gw[kU + c] + cv[j] * gw[2 * kU + c];
  }
  dot = wave_sum(dot) + gb[0];
  float g = 1.0f / (1.0f + __expf(-dot));
  g = 1.0f / (1.0f + __expf(-g));      // double sigmoid, as in source
  float mk = mask[r];
  float x[4];
#pragma unroll
  for (int j = 0; j < 4; ++j) x[j] = (lv[j] * g + gv[j] * (1.0f - g)) * mk;
  float mu = wave_sum(x[0] + x[1] + x[2] + x[3]) * (1.0f / kU);
  float var = 0.f;
#pragma unroll
  for (int j = 0; j < 4; ++j) { float d2 = x[j] - mu; var += d2 * d2; }
  var = wave_sum(var) * (1.0f / kU);
  float inv = rsqrtf(var + kEPS);
  float lg = 0.f;
#pragma unroll
  for (int j = 0; j < 4; ++j) lg += (x[j] - mu) * inv * cv[j];
  lg = wave_sum(lg);
  if (lane == 0) out[row] = lg;
}

// ---------------------------------------------------------------------------
extern "C" void kernel_launch(void* const* d_in, const int* in_sizes, int n_in,
                              void* d_out, int out_size, void* d_ws, size_t ws_size,
                              hipStream_t stream) {
  (void)in_sizes; (void)n_in; (void)out_size; (void)ws_size;
  const float* item_emb  = (const float*)d_in[0];
  const float* pos_emb   = (const float*)d_in[1];
  const float* sab_Wq    = (const float*)d_in[2];
  const float* sab_Wk    = (const float*)d_in[3];
  const float* sab_Wv    = (const float*)d_in[4];
  const float* ff1_W     = (const float*)d_in[5];
  const float* ff1_b     = (const float*)d_in[6];
  const float* ff2_W     = (const float*)d_in[7];
  const float* ff2_b     = (const float*)d_in[8];
  const float* lbab_Q    = (const float*)d_in[9];
  const float* lbab_Wk   = (const float*)d_in[10];
  const float* lbab_Wv   = (const float*)d_in[11];
  const float* lff1_W    = (const float*)d_in[12];
  const float* lff1_b    = (const float*)d_in[13];
  const float* lff2_W    = (const float*)d_in[14];
  const float* lff2_b    = (const float*)d_in[15];
  const float* gated_w   = (const float*)d_in[16];
  const float* gated_b   = (const float*)d_in[17];
  const float* maskp     = (const float*)d_in[18];
  const int*   user_it   = (const int*)d_in[19];
  const int*   pos_it    = (const int*)d_in[20];
  const int*   neg_it    = (const int*)d_in[21];
  float* outp = (float*)d_out;

  // ---- carve workspace ----
  uint8_t* p = (uint8_t*)d_ws;
  auto take = [&](size_t n) -> void* {
    void* q = (void*)p;
    p += (n + 255) & ~(size_t)255;
    return q;
  };
  __bf16* Wqb = (__bf16*)take(sizeof(__bf16) * kU * kU);
  __bf16* Wkb = (__bf16*)take(sizeof(__bf16) * kU * kU);
  __bf16* Wvb = (__bf16*)take(sizeof(__bf16) * kU * kU);
  __bf16* W1b = (__bf16*)take(sizeof(__bf16) * kU * kU);
  __bf16* W2b = (__bf16*)take(sizeof(__bf16) * kU * kU);
  __bf16* Qb  = (__bf16*)take(sizeof(__bf16) * (size_t)kB * kLP * kU);
  __bf16* Ktb = (__bf16*)take(sizeof(__bf16) * (size_t)kB * kU * kLP);
  __bf16* Vb  = (__bf16*)take(sizeof(__bf16) * (size_t)kB * kLP * kU);
  float*  locA = (float*)take(sizeof(float) * (size_t)kB * kL * kU);
  float*  locB = (float*)take(sizeof(float) * (size_t)kB * kL * kU);
  float*  tmp  = (float*)take(sizeof(float) * (size_t)kB * kL * kU);
  float*  rq      = (float*)take(sizeof(float) * 2 * kU);
  float*  glo_raw = (float*)take(sizeof(float) * kB * kU);
  float*  glo     = (float*)take(sizeof(float) * kB * kU);

  const int nrows   = kB * kL;          // 102400
  const int lnGrid  = nrows / 8;        // 12800
  const int cvtGrid = (kU * kU + 255) / 256;

  // bf16 weight copies (L2-resident, reused by every tile)
  k_cvt<<<cvtGrid, 256, 0, stream>>>(sab_Wq, Wqb, kU * kU);
  k_cvt<<<cvtGrid, 256, 0, stream>>>(sab_Wk, Wkb, kU * kU);
  k_cvt<<<cvtGrid, 256, 0, stream>>>(sab_Wv, Wvb, kU * kU);
  k_cvt<<<cvtGrid, 256, 0, stream>>>(ff1_W, W1b, kU * kU);
  k_cvt<<<cvtGrid, 256, 0, stream>>>(ff2_W, W2b, kU * kU);

  // inputs
  k_embed<<<lnGrid, 256, 0, stream>>>(item_emb, pos_emb, user_it, maskp, locA);

  // two shared-weight SAB blocks
  for (int blk = 0; blk < 2; ++blk) {
    k_qkv<<<kB * (kLP / 16), 256, 0, stream>>>(locA, Wqb, Wkb, Wvb, Qb, Ktb, Vb);
    k_attn<<<kB * kH * 13, 32, 0, stream>>>(Qb, Ktb, Vb, locA, maskp, tmp);
    k_ln<<<lnGrid, 256, 0, stream>>>(tmp, locB, nullptr, nrows);
    k_ffn<<<kB * 13, 256, 0, stream>>>(locB, W1b, W2b, ff1_b, ff2_b, locA);
    k_ln<<<lnGrid, 256, 0, stream>>>(locA, locA, maskp, nrows);
  }

  // global branch (algebraically reduced loc-attention)
  k_locq<<<1, 128, 0, stream>>>(lbab_Wk, lbab_Q, rq);
  k_glo_attn<<<kB, 256, 0, stream>>>(item_emb, user_it, maskp, rq, lbab_Wv, glo_raw);
  k_glo_post<<<kB, 128, 0, stream>>>(glo_raw, lff1_W, lff1_b, lff2_W, lff2_b, glo);

  // gating + logits
  k_gate<<<(2 * nrows) / 8, 256, 0, stream>>>(locA, glo, item_emb, user_it, pos_it,
                                              neg_it, gated_w, gated_b, maskp, outp);
}
